// MultiHeadSelfAttention_16355235463384
// MI455X (gfx1250) — matrix operand order repaired and checked by reading, not compiled, not run on hardware
//
#include <hip/hip_runtime.h>
#include <hip/hip_bf16.h>

// ---------------------------------------------------------------------------
// MultiHeadSelfAttention (num/den logits variant) for MI455X / gfx1250.
//  - f16 WMMA (v_wmma_f32_16x16x32_f16) for all matmuls, f32 accumulate.
//  - flash-attention streaming: [B,H,S,S] logits never touch memory.
//  - K / (K+neg) / V tiles staged into LDS by the Tensor Data Mover
//    (tensor_load_to_lds, TENSORcnt, double buffered) -> one DMA issue per
//    tile per block instead of 8 redundant per-wave global load streams.
//  - V stored transposed per head so every WMMA B-fragment is contiguous.
//  - per-wave LDS round-trip converts softmax P from C-layout to A-layout.
// ---------------------------------------------------------------------------

typedef __attribute__((ext_vector_type(16))) _Float16     v16h;
typedef __attribute__((ext_vector_type(8)))  _Float16     v8h;
typedef __attribute__((ext_vector_type(8)))  float        v8f;
typedef __attribute__((ext_vector_type(4)))  unsigned int u32x4;
typedef __attribute__((ext_vector_type(8)))  unsigned int u32x8;

#define D_MODEL   1024
#define NUM_HEADS 16
#define HEAD_DIM  64
#define SEQ       2048
#define BATCH     2
#define NTOK      (BATCH * SEQ)   // 4096

static __device__ __forceinline__ v8f wmma_f16(v16h a, v16h b, v8f c) {
  return __builtin_amdgcn_wmma_f32_16x16x32_f16(
      /*neg_a=*/false, a, /*neg_b=*/false, b,
      /*c_mod=*/(short)0, c, /*reuse_a=*/false, /*reuse_b=*/false);
}

// A-matrix 16x32 f16 fragment from a row-major tile (contiguous k order):
// lane = row (lid&15); hi half of wave selects K {0-15} vs {16-31}.
static __device__ __forceinline__ v16h load_a_frag(const _Float16* __restrict__ base,
                                                   int ld, int row, int hi, int kb) {
  const _Float16* p0 = base + (size_t)row * ld + kb + hi * 16;
  return *(const v16h*)p0;
}

// ---------------------------------------------------------------------------
// Tensor Data Mover: 2-D tile (rows x width f16, row stride in elements) from
// global memory into LDS at lds_byte_off. D# built per ISA ch.8 (group0 128b,
// group1 256b). Issued from a single wave; tracked on TENSORcnt.
// ---------------------------------------------------------------------------
static __device__ __forceinline__ void tdm_load_tile_2d(unsigned lds_byte_off,
                                                        const _Float16* gsrc,
                                                        unsigned width_elems,
                                                        unsigned rows,
                                                        unsigned stride_elems) {
  unsigned long long ga = (unsigned long long)(uintptr_t)gsrc;
  u32x4 g0;
  g0[0] = 1u;                                    // count=1, user descriptor
  g0[1] = lds_byte_off;                          // lds_addr (bytes)
  g0[2] = (unsigned)ga;                          // global_addr[31:0]
  g0[3] = ((unsigned)(ga >> 32) & 0x01FFFFFFu)   // global_addr[56:32]
          | (2u << 30);                          // type = 2 ("image")
  u32x8 g1;
  g1[0] = (1u << 16);                            // data_size=1 (2B); wg_mask=0
  g1[1] = (width_elems << 16);                   // tensor_dim0[15:0]
  g1[2] = (rows << 16);                          // tensor_dim1[15:0]
  g1[3] = (width_elems << 16);                   // tile_dim0
  g1[4] = rows;                                  // tile_dim1 (tile_dim2=0)
  g1[5] = stride_elems;                          // tensor_dim0_stride[31:0]
  g1[6] = 0u;
  g1[7] = 0u;
  asm volatile("tensor_load_to_lds %0, %1" :: "s"(g0), "s"(g1) : "memory");
}

// ---------------------------------------------------------------------------
// elementwise helpers
// ---------------------------------------------------------------------------
__global__ void cvt_f32_f16_kernel(const float* __restrict__ s,
                                   _Float16* __restrict__ d, int n) {
  int i = blockIdx.x * blockDim.x + threadIdx.x;
  if (i < n) d[i] = (_Float16)s[i];
}

__global__ void add_f16_kernel(const _Float16* __restrict__ a,
                               const _Float16* __restrict__ b,
                               _Float16* __restrict__ c, int n) {
  int i = blockIdx.x * blockDim.x + threadIdx.x;
  if (i < n) c[i] = (_Float16)((float)a[i] + (float)b[i]);
}

// ---------------------------------------------------------------------------
// Fused projection GEMM: out = act @ W^T + b for {Q,K,V,neg} (blockIdx.z).
// 8 waves x 32(M) rows = 256-row block tile; 64(N); K step 32.
// 2 A-frags reused across 4 B-columns -> 8 WMMAs / 6 loads per k-step.
// V is written transposed per head: Vt[b,h,d,s].
// ---------------------------------------------------------------------------
__global__ __launch_bounds__(256)
void proj_kernel(const _Float16* __restrict__ q16, const _Float16* __restrict__ kin16,
                 const _Float16* __restrict__ Wq, const _Float16* __restrict__ Wk,
                 const _Float16* __restrict__ Wv, const _Float16* __restrict__ Wn,
                 const float* __restrict__ bq, const float* __restrict__ bk,
                 const float* __restrict__ bv, const float* __restrict__ bn,
                 _Float16* __restrict__ Qb, _Float16* __restrict__ Kb,
                 _Float16* __restrict__ Vt, _Float16* __restrict__ Nb) {
  const int pid = blockIdx.z;
  const _Float16* act  = (pid == 3) ? kin16 : q16;
  const _Float16* W    = (pid == 0) ? Wq : (pid == 1) ? Wk : (pid == 2) ? Wv : Wn;
  const float*    bias = (pid == 0) ? bq : (pid == 1) ? bk : (pid == 2) ? bv : bn;

  const int wave = threadIdx.x >> 5;
  const int lid  = threadIdx.x & 31;
  const int row  = lid & 15;       // A row / B column within tile
  const int hi   = lid >> 4;
  const int mBase = blockIdx.x * 256 + wave * 32;
  const int nBase = blockIdx.y * 64;

  v8f acc[2][4] = {};
  const _Float16* aBase = act + (size_t)mBase * D_MODEL;

  for (int k0 = 0; k0 < D_MODEL; k0 += 32) {
    v16h a0 = load_a_frag(aBase,                D_MODEL, row, hi, k0);
    v16h a1 = load_a_frag(aBase + 16 * D_MODEL, D_MODEL, row, hi, k0);
#pragma unroll
    for (int t = 0; t < 4; ++t) {
      // B[k,n] = W[n,k]: lane n reads 16 contiguous f16 from W row n.
      v16h b = *(const v16h*)(W + (size_t)(nBase + t * 16 + row) * D_MODEL + k0 + hi * 16);
      acc[0][t] = wmma_f16(a0, b, acc[0][t]);
      acc[1][t] = wmma_f16(a1, b, acc[1][t]);
    }
  }

#pragma unroll
  for (int g = 0; g < 2; ++g) {
    const int mB = mBase + g * 16;
    if (pid != 2) {
      _Float16* outp = (pid == 0) ? Qb : (pid == 1) ? Kb : Nb;
#pragma unroll
      for (int t = 0; t < 4; ++t) {
        const int nG = nBase + t * 16 + row;
        const float bb = bias[nG];
#pragma unroll
        for (int r = 0; r < 8; ++r) {
          const int mG = mB + 8 * hi + r;   // C-layout: VGPR r -> row r (+8 hi)
          outp[(size_t)mG * D_MODEL + nG] = (_Float16)(acc[g][t][r] + bb);
        }
      }
    } else {
      // Transposed per-head store: Vt[((b*H+h)*64+d)*SEQ + s]; 8 C-rows are
      // 8 consecutive s values -> one packed v8h store per tile per lane.
      const int mG0 = mB + 8 * hi;
      const int bIx = mG0 / SEQ;
      const int s0  = mG0 % SEQ;
#pragma unroll
      for (int t = 0; t < 4; ++t) {
        const int nG = nBase + t * 16 + row;
        const int h  = nG >> 6, dcol = nG & 63;
        const float bb = bias[nG];
        v8h pk;
#pragma unroll
        for (int r = 0; r < 8; ++r) pk[r] = (_Float16)(acc[g][t][r] + bb);
        *(v8h*)(Vt + ((size_t)(bIx * NUM_HEADS + h) * HEAD_DIM + dcol) * SEQ + s0) = pk;
      }
    }
  }
}

// ---------------------------------------------------------------------------
// Flash attention: per (b,h), 8 waves x 16 q-rows = 128-row q-tile per block.
// K/Kn/V 64x64 tiles TDM-staged into LDS, double buffered. Per k-tile:
// num = Q K^T, den = K (K+neg)^T, logits = num/den, online softmax,
// P -> LDS -> A-frags, O += P V.
// ---------------------------------------------------------------------------
__global__ __launch_bounds__(256)
void attn_kernel(const _Float16* __restrict__ Qb, const _Float16* __restrict__ Kb,
                 const _Float16* __restrict__ Knb, const _Float16* __restrict__ Vt,
                 _Float16* __restrict__ Ob) {
  __shared__ _Float16 ldsK[2][64 * 64];   // K  tile  [s][d], 8 KB x2
  __shared__ _Float16 ldsN[2][64 * 64];   // Kn tile  [s][d]
  __shared__ _Float16 ldsV[2][64 * 64];   // V  tile  [d][s] (from Vt)
  __shared__ _Float16 ldsP[8][16 * 64];   // per-wave P scratch, 16 KB

  const int wave = threadIdx.x >> 5;
  const int lid  = threadIdx.x & 31;
  const int row  = lid & 15;
  const int hi   = lid >> 4;
  const int bh = blockIdx.y;
  const int b  = bh >> 4;
  const int h  = bh & 15;
  const int qw = blockIdx.x * 128 + wave * 16;

  const _Float16* kbase  = Kb  + ((size_t)b * SEQ) * D_MODEL + h * HEAD_DIM;
  const _Float16* knbase = Knb + ((size_t)b * SEQ) * D_MODEL + h * HEAD_DIM;
  const _Float16* vtbase = Vt + ((size_t)(b * NUM_HEADS + h) * HEAD_DIM) * SEQ;

  // Left operands, loaded once: Q rows and K rows at the q positions.
  const _Float16* qbase  = Qb + ((size_t)(b * SEQ + qw)) * D_MODEL + h * HEAD_DIM;
  const _Float16* kqbase = kbase + (size_t)qw * D_MODEL;
  v16h aQ[2], aK[2];
#pragma unroll
  for (int kc = 0; kc < 2; ++kc) {
    aQ[kc] = load_a_frag(qbase,  D_MODEL, row, hi, kc * 32);
    aK[kc] = load_a_frag(kqbase, D_MODEL, row, hi, kc * 32);
  }

  v8f acc[4] = {};               // O accumulator: 16 q x 64 d
  float m_run[8], l_run[8];
#pragma unroll
  for (int r = 0; r < 8; ++r) { m_run[r] = -3.0e38f; l_run[r] = 0.0f; }

  _Float16* pbuf = &ldsP[wave][0];

  // TDM preload of k-tile 0 (one wave issues; EXEC-independent; TENSORcnt).
  if (wave == 0) {
    tdm_load_tile_2d((unsigned)(uintptr_t)&ldsK[0][0], kbase,  64, 64, D_MODEL);
    tdm_load_tile_2d((unsigned)(uintptr_t)&ldsN[0][0], knbase, 64, 64, D_MODEL);
    tdm_load_tile_2d((unsigned)(uintptr_t)&ldsV[0][0], vtbase, 64, 64, SEQ);
    __builtin_amdgcn_s_wait_tensorcnt(0);
  }
  __syncthreads();

  for (int kt = 0; kt < SEQ; kt += 64) {
    const int cur = (kt >> 6) & 1;
    const int nxt = cur ^ 1;

    // Kick off next tile's DMA before computing on the current one.
    if (wave == 0 && (kt + 64) < SEQ) {
      tdm_load_tile_2d((unsigned)(uintptr_t)&ldsK[nxt][0],
                       kbase  + (size_t)(kt + 64) * D_MODEL, 64, 64, D_MODEL);
      tdm_load_tile_2d((unsigned)(uintptr_t)&ldsN[nxt][0],
                       knbase + (size_t)(kt + 64) * D_MODEL, 64, 64, D_MODEL);
      tdm_load_tile_2d((unsigned)(uintptr_t)&ldsV[nxt][0],
                       vtbase + (kt + 64), 64, 64, SEQ);
    }

    const _Float16* kl = &ldsK[cur][0];
    const _Float16* nl = &ldsN[cur][0];
    const _Float16* vl = &ldsV[cur][0];

    v8f lg[4];
#pragma unroll
    for (int t = 0; t < 4; ++t) {
      v8f num = {}; v8f den = {};
#pragma unroll
      for (int kc = 0; kc < 2; ++kc) {
        // num B[d,n] = K[kt+n, d]: contiguous slice of LDS K-tile row n.
        v16h bK = *(const v16h*)(kl + (t * 16 + row) * 64 + kc * 32 + hi * 16);
        num = wmma_f16(aQ[kc], bK, num);
      }
#pragma unroll
      for (int kc = 0; kc < 2; ++kc) {
        v16h bN = *(const v16h*)(nl + (t * 16 + row) * 64 + kc * 32 + hi * 16);
        den = wmma_f16(aK[kc], bN, den);
      }
#pragma unroll
      for (int r = 0; r < 8; ++r) lg[t][r] = num[r] / den[r];
    }

    // Online softmax. Row M lives across the 16 lanes of one half-wave;
    // shfl_xor {8,4,2,1} reduces within that group.
#pragma unroll
    for (int r = 0; r < 8; ++r) {
      float lm = fmaxf(fmaxf(lg[0][r], lg[1][r]), fmaxf(lg[2][r], lg[3][r]));
#pragma unroll
      for (int off = 8; off >= 1; off >>= 1) lm = fmaxf(lm, __shfl_xor(lm, off, 32));
      const float mn = fmaxf(m_run[r], lm);
      const float sc = __expf(m_run[r] - mn);
      m_run[r] = mn;
      float rs = 0.0f;
#pragma unroll
      for (int t = 0; t < 4; ++t) {
        const float p = __expf(lg[t][r] - mn);
        lg[t][r] = p;
        rs += p;
      }
#pragma unroll
      for (int off = 8; off >= 1; off >>= 1) rs += __shfl_xor(rs, off, 32);
      l_run[r] = l_run[r] * sc + rs;
#pragma unroll
      for (int t2 = 0; t2 < 4; ++t2) acc[t2][r] *= sc;
    }

    // P (C-layout f32) -> LDS f16 [16][64]. Per-wave region; LDS ops from one
    // wave stay in order, so no barrier needed before the re-load.
#pragma unroll
    for (int t = 0; t < 4; ++t)
#pragma unroll
      for (int r = 0; r < 8; ++r)
        pbuf[(8 * hi + r) * 64 + t * 16 + row] = (_Float16)lg[t][r];

    // O += P V  (K-dim = 64 k-positions -> 2 chunks of 32).
#pragma unroll
    for (int kc = 0; kc < 2; ++kc) {
      v16h aP = load_a_frag(pbuf, 64, row, hi, kc * 32);
#pragma unroll
      for (int t2 = 0; t2 < 4; ++t2) {
        // B[kp,n] = V[kt+kp, d=n] = LDS V-tile row d, contiguous along s.
        v16h bV = *(const v16h*)(vl + (t2 * 16 + row) * 64 + kc * 32 + hi * 16);
        acc[t2] = wmma_f16(aP, bV, acc[t2]);
      }
    }

    // Next tile's DMA must be complete (wave0 waits TENSORcnt) and all waves
    // must be done reading `cur` before it is overwritten next iteration.
    if ((kt + 64) < SEQ && wave == 0) __builtin_amdgcn_s_wait_tensorcnt(0);
    __syncthreads();
  }

  // Normalize and write attention output (f16, row-major merged heads).
#pragma unroll
  for (int t2 = 0; t2 < 4; ++t2) {
    const int nG = h * HEAD_DIM + t2 * 16 + row;
#pragma unroll
    for (int r = 0; r < 8; ++r) {
      const int tok = b * SEQ + qw + 8 * hi + r;
      Ob[(size_t)tok * D_MODEL + nG] = (_Float16)(acc[t2][r] / l_run[r]);
    }
  }
}

// ---------------------------------------------------------------------------
// Output projection: out_f32 = attn16 @ Wo^T + bo. Same 32-row wave tiling.
// ---------------------------------------------------------------------------
__global__ __launch_bounds__(256)
void outproj_kernel(const _Float16* __restrict__ A, const _Float16* __restrict__ Wo,
                    const float* __restrict__ bo, float* __restrict__ out) {
  const int wave = threadIdx.x >> 5;
  const int lid  = threadIdx.x & 31;
  const int row  = lid & 15;
  const int hi   = lid >> 4;
  const int mBase = blockIdx.x * 256 + wave * 32;
  const int nBase = blockIdx.y * 64;

  v8f acc[2][4] = {};
  const _Float16* aBase = A + (size_t)mBase * D_MODEL;
  for (int k0 = 0; k0 < D_MODEL; k0 += 32) {
    v16h a0 = load_a_frag(aBase,                D_MODEL, row, hi, k0);
    v16h a1 = load_a_frag(aBase + 16 * D_MODEL, D_MODEL, row, hi, k0);
#pragma unroll
    for (int t = 0; t < 4; ++t) {
      v16h bf = *(const v16h*)(Wo + (size_t)(nBase + t * 16 + row) * D_MODEL + k0 + hi * 16);
      acc[0][t] = wmma_f16(a0, bf, acc[0][t]);
      acc[1][t] = wmma_f16(a1, bf, acc[1][t]);
    }
  }
#pragma unroll
  for (int g = 0; g < 2; ++g) {
    const int mB = mBase + g * 16;
#pragma unroll
    for (int t = 0; t < 4; ++t) {
      const int nG = nBase + t * 16 + row;
      const float bb = bo[nG];
#pragma unroll
      for (int r = 0; r < 8; ++r) {
        const int mG = mB + 8 * hi + r;
        out[(size_t)mG * D_MODEL + nG] = acc[g][t][r] + bb;
      }
    }
  }
}

// ---------------------------------------------------------------------------
// Host-side launch. Inputs (setup_inputs order):
// 0 query 1 key 2 Wq 3 bq 4 Wk 5 bk 6 Wv 7 bv 8 Wneg 9 bneg 10 Wo 11 bo
// ---------------------------------------------------------------------------
extern "C" void kernel_launch(void* const* d_in, const int* in_sizes, int n_in,
                              void* d_out, int out_size, void* d_ws, size_t ws_size,
                              hipStream_t stream) {
  (void)in_sizes; (void)n_in; (void)out_size; (void)ws_size;

  const float* query = (const float*)d_in[0];
  const float* key   = (const float*)d_in[1];
  const float* Wq = (const float*)d_in[2];  const float* bq = (const float*)d_in[3];
  const float* Wk = (const float*)d_in[4];  const float* bk = (const float*)d_in[5];
  const float* Wv = (const float*)d_in[6];  const float* bv = (const float*)d_in[7];
  const float* Wn = (const float*)d_in[8];  const float* bn = (const float*)d_in[9];
  const float* Wo = (const float*)d_in[10]; const float* bo = (const float*)d_in[11];
  float* out = (float*)d_out;

  const size_t nAct = (size_t)NTOK * D_MODEL;      // 4M elems
  const size_t nW   = (size_t)D_MODEL * D_MODEL;   // 1M elems
  char* ws = (char*)d_ws;
  size_t off = 0;
  auto carve = [&](size_t elems) {
    _Float16* p = (_Float16*)(ws + off);
    off += ((elems * sizeof(_Float16)) + 255) & ~(size_t)255;
    return p;
  };
  _Float16* q16  = carve(nAct);
  _Float16* k16  = carve(nAct);
  _Float16* Wq16 = carve(nW);
  _Float16* Wk16 = carve(nW);
  _Float16* Wv16 = carve(nW);
  _Float16* Wn16 = carve(nW);
  _Float16* Wo16 = carve(nW);
  _Float16* Qb   = carve(nAct);
  _Float16* Kb   = carve(nAct);
  _Float16* Nb   = carve(nAct);
  _Float16* Knb  = carve(nAct);
  _Float16* Vt   = carve(nAct);   // transposed per head [b,h,d,s]
  _Float16* Ob   = carve(nAct);   // total ~74 MB

  const int TB = 256;
  // f32 -> f16 staging
  cvt_f32_f16_kernel<<<(int)((nAct + TB - 1) / TB), TB, 0, stream>>>(query, q16, (int)nAct);
  cvt_f32_f16_kernel<<<(int)((nAct + TB - 1) / TB), TB, 0, stream>>>(key,   k16, (int)nAct);
  cvt_f32_f16_kernel<<<(int)((nW + TB - 1) / TB), TB, 0, stream>>>(Wq, Wq16, (int)nW);
  cvt_f32_f16_kernel<<<(int)((nW + TB - 1) / TB), TB, 0, stream>>>(Wk, Wk16, (int)nW);
  cvt_f32_f16_kernel<<<(int)((nW + TB - 1) / TB), TB, 0, stream>>>(Wv, Wv16, (int)nW);
  cvt_f32_f16_kernel<<<(int)((nW + TB - 1) / TB), TB, 0, stream>>>(Wn, Wn16, (int)nW);
  cvt_f32_f16_kernel<<<(int)((nW + TB - 1) / TB), TB, 0, stream>>>(Wo, Wo16, (int)nW);

  // Q,K,V,neg projections (V stored transposed per head)
  proj_kernel<<<dim3(NTOK / 256, D_MODEL / 64, 4), TB, 0, stream>>>(
      q16, k16, Wq16, Wk16, Wv16, Wn16, bq, bk, bv, bn, Qb, Kb, Vt, Nb);

  // Kn = K + neg
  add_f16_kernel<<<(int)((nAct + TB - 1) / TB), TB, 0, stream>>>(Kb, Nb, Knb, (int)nAct);

  // Flash attention with TDM-staged tiles
  attn_kernel<<<dim3(SEQ / 128, BATCH * NUM_HEADS), TB, 0, stream>>>(Qb, Kb, Knb, Vt, Ob);

  // Output projection
  outproj_kernel<<<dim3(NTOK / 256, D_MODEL / 64), TB, 0, stream>>>(Ob, Wo16, bo, out);
}
